// EncoderBlock_11020886081577
// MI455X (gfx1250) — matrix operand order repaired and checked by reading, not compiled, not run on hardware
//
#include <hip/hip_runtime.h>
#include <hip/hip_bf16.h>

typedef __attribute__((ext_vector_type(16))) _Float16 v16h;
typedef __attribute__((ext_vector_type(8)))  float    v8f;

union Frag16 { v16h v; _Float16 h[16]; unsigned u[8]; };
union H4 { _Float16 h[4]; unsigned u[2]; };

#define NB 8
#define NC 128
#define NO 256
#define NH 128
#define NW 128
#define NHW 16384
#define NHEADS 8
#define KSPLIT 32

// workspace offsets (in floats)
#define OFF_DW    0                    // 8*128*16384
#define OFF_G     16777216             // 8*128*128
#define OFF_BNSUM 16908288             // 128
#define OFF_BNSQ  16908416             // 128
#define OFF_SX    16908544             // 8*128
#define OFF_M     16909568             // 8*128*128
#define OFF_CV    17040640             // 8*128
#define OFF_WP    17041664             // 256*128
#define OFF_BP    17074432             // 256

static __device__ inline v8f vzero8() {
    v8f z;
#pragma unroll
    for (int i = 0; i < 8; i++) z[i] = 0.f;
    return z;
}

__global__ void k_zero(float* p, int n) {
    int i = blockIdx.x * 256 + threadIdx.x;
    if (i < n) p[i] = 0.f;
}

// ---------------------------------------------------------------------------
// K1: axial depthwise conv (7x1 + 1x7, zero pad) + BN partial stats + row sums
// grid = B*C, one (b,c) plane per workgroup; float4 in/out
// ---------------------------------------------------------------------------
__global__ __launch_bounds__(256) void k_axial(
    const float* __restrict__ x,
    const float* __restrict__ whv, const float* __restrict__ bhv,
    const float* __restrict__ wwv, const float* __restrict__ bwv,
    float* __restrict__ dw, float* __restrict__ bn_sum, float* __restrict__ bn_sq,
    float* __restrict__ sum_x)
{
    __shared__ __align__(16) float tile[NHW];   // 64 KB plane
    __shared__ float red[3][8];
    int bc = blockIdx.x;
    int c = bc & 127;
    const float4* xp4 = (const float4*)(x + (size_t)bc * NHW);
    float4* tile4 = (float4*)tile;
    float sx = 0.f;
    for (int i = threadIdx.x; i < NHW / 4; i += 256) {
        float4 v = xp4[i];
        tile4[i] = v;
        sx += v.x + v.y + v.z + v.w;
    }
    float wv_[7], wh_[7];
#pragma unroll
    for (int k = 0; k < 7; k++) { wv_[k] = whv[c * 7 + k]; wh_[k] = wwv[c * 7 + k]; }
    float bias = bhv[c] + bwv[c];
    __syncthreads();
    float s = 0.f, s2 = 0.f;
    float* dwp = dw + (size_t)bc * NHW;
    for (int j = 0; j < 16; j++) {
        int i4 = threadIdx.x * 4 + j * 1024;
        int h = i4 >> 7, w0 = i4 & 127;
        float o4[4];
#pragma unroll
        for (int l = 0; l < 4; l++) {
            int w = w0 + l;
            float acc = bias;
#pragma unroll
            for (int k = 0; k < 7; k++) { int hh = h + k - 3; if (hh >= 0 && hh < NH) acc += wv_[k] * tile[(hh << 7) | w]; }
#pragma unroll
            for (int k = 0; k < 7; k++) { int w2 = w + k - 3; if (w2 >= 0 && w2 < NW) acc += wh_[k] * tile[(h << 7) | w2]; }
            o4[l] = acc;
            s += acc; s2 += acc * acc;
        }
        float4 st; st.x = o4[0]; st.y = o4[1]; st.z = o4[2]; st.w = o4[3];
        *(float4*)&dwp[i4] = st;
    }
    for (int off = 16; off > 0; off >>= 1) {
        s  += __shfl_xor(s, off);
        s2 += __shfl_xor(s2, off);
        sx += __shfl_xor(sx, off);
    }
    int lane = threadIdx.x & 31, wave = threadIdx.x >> 5;
    if (lane == 0) { red[0][wave] = s; red[1][wave] = s2; red[2][wave] = sx; }
    __syncthreads();
    if (threadIdx.x == 0) {
        float a = 0.f, b2 = 0.f, c2 = 0.f;
        for (int w2 = 0; w2 < 8; w2++) { a += red[0][w2]; b2 += red[1][w2]; c2 += red[2][w2]; }
        atomicAdd(&bn_sum[c], a);
        atomicAdd(&bn_sq[c], b2);
        sum_x[bc] = c2;
    }
}

// ---------------------------------------------------------------------------
// K2: per-batch Gram matrix G[b] = X[b] X[b]^T via f16 WMMA, split-K + atomics
// grid = B*KSPLIT; each wg covers 512 positions; 8 waves x 8 n-tiles
// ---------------------------------------------------------------------------
__global__ __launch_bounds__(256) void k_gram(const float* __restrict__ x, float* __restrict__ G)
{
    __shared__ __align__(16) _Float16 xs[NC * 32];   // [channel][k] 8 KB
    int wg = blockIdx.x;
    int b = wg >> 5;
    int ks = wg & 31;
    int kbase0 = ks * (NHW / KSPLIT);
    int tid = threadIdx.x, lane = tid & 31, wave = tid >> 5;
    const float* xb = x + (size_t)b * NC * NHW;
    v8f acc[8];
#pragma unroll
    for (int nt = 0; nt < 8; nt++) acc[nt] = vzero8();

    for (int kc = 0; kc < NHW / KSPLIT; kc += 32) {
        int kb = kbase0 + kc;
        __syncthreads();
        // float4 global load, pack 4 halfs, 8B LDS store
        for (int i = tid; i < NC * 8; i += 256) {
            int c2 = i >> 3, k4 = (i & 7) * 4;
            float4 v = *(const float4*)&xb[(size_t)c2 * NHW + kb + k4];
            H4 t;
            t.h[0] = (_Float16)v.x; t.h[1] = (_Float16)v.y;
            t.h[2] = (_Float16)v.z; t.h[3] = (_Float16)v.w;
            *(H4*)&xs[c2 * 32 + k4] = t;
        }
        __syncthreads();
        Frag16 a;
        {
            int m = wave * 16 + (lane & 15);
            int kh = (lane >> 4) * 16;
#pragma unroll
            for (int v = 0; v < 8; v++) {
                int kk = kh + v * 2;
                a.u[v] = *(const unsigned*)&xs[m * 32 + kk];
            }
        }
        Frag16 bfr[8];
#pragma unroll
        for (int nt = 0; nt < 8; nt++) {
            int nn = nt * 16 + (lane & 15);
            int kb2 = (lane >> 4) * 16;
#pragma unroll
            for (int v = 0; v < 8; v++) bfr[nt].u[v] = *(const unsigned*)&xs[nn * 32 + kb2 + 2 * v];
        }
#pragma unroll
        for (int nt = 0; nt < 8; nt++)
            acc[nt] = __builtin_amdgcn_wmma_f32_16x16x32_f16(
                false, a.v, false, bfr[nt].v, (short)0, acc[nt], false, false);
    }
    float* Gbp = G + (size_t)b * NC * NC;
#pragma unroll
    for (int nt = 0; nt < 8; nt++)
#pragma unroll
        for (int v = 0; v < 8; v++) {
            int row = wave * 16 + v + 8 * (lane >> 4);
            int col = nt * 16 + (lane & 15);
            atomicAdd(&Gbp[row * NC + col], acc[nt][v]);
        }
}

// ---------------------------------------------------------------------------
// K3a: BN finalize + fold BN into pointwise weights: W' = pw_w*alpha, b' fold
// ---------------------------------------------------------------------------
__global__ __launch_bounds__(256) void k_bnfold(
    const float* __restrict__ bn_sum, const float* __restrict__ bn_sq,
    const float* __restrict__ gamma, const float* __restrict__ beta,
    const float* __restrict__ pw_w, const float* __restrict__ pw_b,
    float* __restrict__ Wp, float* __restrict__ bp)
{
    __shared__ float alpha[NC], beta2[NC];
    int t = threadIdx.x;
    if (t < NC) {
        float n = (float)(NB * NHW);
        float mean = bn_sum[t] / n;
        float var = bn_sq[t] / n - mean * mean;
        float inv = rsqrtf(var + 1e-5f);
        float a = gamma[t] * inv;
        alpha[t] = a;
        beta2[t] = beta[t] - mean * a;
    }
    __syncthreads();
    float accb = pw_b[t];
    for (int c = 0; c < NC; c++) {
        float w = pw_w[t * NC + c];
        Wp[t * NC + c] = w * alpha[c];
        accb += w * beta2[c];
    }
    bp[t] = accb;
}

// ---------------------------------------------------------------------------
// K3b: per-batch attention math (tiny): scores from Gram + bias cross terms,
// softmax, M[b] = Wo*A_blk*Wv, c[b] = Wo*A_blk*bv + bo.  grid = B
// ---------------------------------------------------------------------------
__global__ __launch_bounds__(256) void k_attn(
    const float* __restrict__ G, const float* __restrict__ sum_x,
    const float* __restrict__ wq, const float* __restrict__ bq,
    const float* __restrict__ wk, const float* __restrict__ bk,
    const float* __restrict__ wv, const float* __restrict__ bv,
    const float* __restrict__ wo, const float* __restrict__ bo,
    float* __restrict__ Mout, float* __restrict__ cvec)
{
    __shared__ float g[NC * NC];          // Gram, later reused for P
    __shared__ float t2[NC * 16];         // G @ Wk_head^T slice
    __shared__ float sc[NHEADS * 16 * 16];
    __shared__ float u1[NC], w2v[NC], pv[NC], sxs[NC];
    int b = blockIdx.x, tid = threadIdx.x;
    const float* Gb = G + (size_t)b * NC * NC;
    for (int i = tid; i < NC * NC; i += 256) g[i] = Gb[i];
    if (tid < NC) sxs[tid] = sum_x[b * NC + tid];
    __syncthreads();
    if (tid < NC) {
        float a = 0.f, bb = 0.f;
        for (int c2 = 0; c2 < NC; c2++) { a += wq[tid * NC + c2] * sxs[c2]; bb += wk[tid * NC + c2] * sxs[c2]; }
        u1[tid] = a; w2v[tid] = bb;
    }
    __syncthreads();
    for (int n = 0; n < NHEADS; n++) {
        for (int i = tid; i < NC * 16; i += 256) {
            int c2 = i >> 4, k = i & 15;
            const float* wkr = wk + (n * 16 + k) * NC;
            float s = 0.f;
            for (int d = 0; d < NC; d++) s += g[c2 * NC + d] * wkr[d];
            t2[i] = s;
        }
        __syncthreads();
        {
            int q = tid >> 4, k = tid & 15;
            const float* wqr = wq + (n * 16 + q) * NC;
            float s = 0.f;
            for (int c2 = 0; c2 < NC; c2++) s += wqr[c2] * t2[c2 * 16 + k];
            int r = n * 16 + q, r2 = n * 16 + k;
            s += u1[r] * bk[r2] + bq[r] * w2v[r2] + (float)NHW * bq[r] * bk[r2];
            sc[n * 256 + tid] = s * 0.25f;   // scale = dh^-0.5 = 1/4
        }
        __syncthreads();
    }
    if (tid < 128) {   // softmax over 16-wide rows
        float* row = sc + tid * 16;
        float mx = row[0];
        for (int k = 1; k < 16; k++) mx = fmaxf(mx, row[k]);
        float ss = 0.f;
        for (int k = 0; k < 16; k++) { float e = expf(row[k] - mx); row[k] = e; ss += e; }
        float inv = 1.f / ss;
        for (int k = 0; k < 16; k++) row[k] *= inv;
    }
    __syncthreads();
    // P = A_blk @ Wv  (overwrite g)
    for (int i = tid; i < NC * NC; i += 256) {
        int r = i >> 7, c2 = i & 127;
        const float* arow = sc + r * 16;
        int n = r >> 4;
        float s = 0.f;
        for (int k = 0; k < 16; k++) s += arow[k] * wv[(n * 16 + k) * NC + c2];
        g[i] = s;
    }
    __syncthreads();
    float* Mb = Mout + (size_t)b * NC * NC;
    for (int i = tid; i < NC * NC; i += 256) {
        int o = i >> 7, c2 = i & 127;
        float s = 0.f;
        for (int r = 0; r < NC; r++) s += wo[o * NC + r] * g[r * NC + c2];
        Mb[i] = s;
    }
    if (tid < NC) {
        const float* arow = sc + tid * 16;
        int n = tid >> 4;
        float s = 0.f;
        for (int k = 0; k < 16; k++) s += arow[k] * bv[n * 16 + k];
        pv[tid] = s;
    }
    __syncthreads();
    if (tid < NC) {
        float s = bo[tid];
        for (int r = 0; r < NC; r++) s += wo[tid * NC + r] * pv[r];
        cvec[b * NC + tid] = s;
    }
}

// ---------------------------------------------------------------------------
// K4: skip0 = M[b] @ X[b] + c[b]  (WMMA f16, K=128)
// grid = B * (HW/128); wg tile = 128 rows x 128 positions; wave = 1 m-tile
// ---------------------------------------------------------------------------
__global__ __launch_bounds__(256) void k_skip0(
    const float* __restrict__ x, const float* __restrict__ Mmat,
    const float* __restrict__ cvec, float* __restrict__ skip0)
{
    __shared__ __align__(16) _Float16 xsT[128 * 36];   // [pos][ch] transposed
    int gid = blockIdx.x;
    int b = gid >> 7;
    int p0 = (gid & 127) * 128;
    int tid = threadIdx.x, lane = tid & 31, wave = tid >> 5;
    const float* xb = x + (size_t)b * NC * NHW;
    const float* Mb = Mmat + (size_t)b * NC * NC;

    Frag16 afr[4];
    int mrow = 16 * wave + (lane & 15);
    int khalf = (lane >> 4) * 16;
#pragma unroll
    for (int kc = 0; kc < 4; kc++)
#pragma unroll
        for (int v = 0; v < 8; v++) {
            int kk = kc * 32 + khalf + v * 2;
            afr[kc].h[2 * v]     = (_Float16)Mb[mrow * NC + kk];
            afr[kc].h[2 * v + 1] = (_Float16)Mb[mrow * NC + kk + 1];
        }

    v8f acc[8];
#pragma unroll
    for (int nt = 0; nt < 8; nt++) acc[nt] = vzero8();

    for (int kc = 0; kc < 4; kc++) {
        __syncthreads();
        // b128 global loads over positions; transposed b16 LDS scatter
        for (int i = tid; i < 32 * 32; i += 256) {
            int cl = i >> 5, p4 = (i & 31) * 4;
            float4 v = *(const float4*)&xb[(size_t)(kc * 32 + cl) * NHW + p0 + p4];
            xsT[(p4 + 0) * 36 + cl] = (_Float16)v.x;
            xsT[(p4 + 1) * 36 + cl] = (_Float16)v.y;
            xsT[(p4 + 2) * 36 + cl] = (_Float16)v.z;
            xsT[(p4 + 3) * 36 + cl] = (_Float16)v.w;
        }
        __syncthreads();
        Frag16 bfr[8];
#pragma unroll
        for (int nt = 0; nt < 8; nt++) {
            int p = nt * 16 + (lane & 15);
            int kb = (lane >> 4) * 16;
#pragma unroll
            for (int v = 0; v < 8; v++) bfr[nt].u[v] = *(const unsigned*)&xsT[p * 36 + kb + 2 * v];
        }
#pragma unroll
        for (int nt = 0; nt < 8; nt++)
            acc[nt] = __builtin_amdgcn_wmma_f32_16x16x32_f16(
                false, afr[kc].v, false, bfr[nt].v, (short)0, acc[nt], false, false);
    }
    float cv8[8];
#pragma unroll
    for (int v = 0; v < 8; v++) cv8[v] = cvec[b * NC + 16 * wave + v + 8 * (lane >> 4)];
#pragma unroll
    for (int nt = 0; nt < 8; nt++)
#pragma unroll
        for (int v = 0; v < 8; v++) {
            int o = 16 * wave + v + 8 * (lane >> 4);
            int p = p0 + nt * 16 + (lane & 15);
            skip0[((size_t)b * NC + o) * NHW + p] = acc[nt][v] + cv8[v];
        }
}

// ---------------------------------------------------------------------------
// K5: pw = W' @ dw + b' (BN folded), fused 2x2 maxpool + exact GELU
// grid = B * 64 row-pairs * 2 col strips; wg tile = 256 rows x (2x64) positions
// ---------------------------------------------------------------------------
__global__ __launch_bounds__(256) void k_pwpool(
    const float* __restrict__ dw, const float* __restrict__ Wp,
    const float* __restrict__ bp, float* __restrict__ out)
{
    __shared__ __align__(16) _Float16 xsT[128 * 36];   // [pos][ch] transposed
    int gid = blockIdx.x;
    int b = gid >> 7;
    int rem = gid & 127;
    int h0 = rem >> 1;
    int w0 = (rem & 1) * 64;
    int tid = threadIdx.x, lane = tid & 31, wave = tid >> 5;
    const float* dwb = dw + (size_t)b * NC * NHW;

    Frag16 afr[2][4];
    int khalf = (lane >> 4) * 16;
#pragma unroll
    for (int mt = 0; mt < 2; mt++) {
        int row = 32 * wave + 16 * mt + (lane & 15);
#pragma unroll
        for (int kc = 0; kc < 4; kc++)
#pragma unroll
            for (int v = 0; v < 8; v++) {
                int kk = kc * 32 + khalf + v * 2;
                afr[mt][kc].h[2 * v]     = (_Float16)Wp[row * NC + kk];
                afr[mt][kc].h[2 * v + 1] = (_Float16)Wp[row * NC + kk + 1];
            }
    }
    v8f acc[2][8];
#pragma unroll
    for (int mt = 0; mt < 2; mt++)
#pragma unroll
        for (int nt = 0; nt < 8; nt++) acc[mt][nt] = vzero8();

    for (int kc = 0; kc < 4; kc++) {
        __syncthreads();
        for (int i = tid; i < 32 * 32; i += 256) {
            int cl = i >> 5, j4 = (i & 31) * 4;
            int r = j4 >> 6, col = j4 & 63;
            int p = (2 * h0 + r) * NW + w0 + col;
            float4 v = *(const float4*)&dwb[(size_t)(kc * 32 + cl) * NHW + p];
            xsT[(j4 + 0) * 36 + cl] = (_Float16)v.x;
            xsT[(j4 + 1) * 36 + cl] = (_Float16)v.y;
            xsT[(j4 + 2) * 36 + cl] = (_Float16)v.z;
            xsT[(j4 + 3) * 36 + cl] = (_Float16)v.w;
        }
        __syncthreads();
#pragma unroll
        for (int nt = 0; nt < 8; nt++) {
            Frag16 bf;
            int j = ((nt >> 2) << 6) | ((nt & 3) << 4) | (lane & 15);
            int kb = (lane >> 4) * 16;
#pragma unroll
            for (int v = 0; v < 8; v++) bf.u[v] = *(const unsigned*)&xsT[j * 36 + kb + 2 * v];
#pragma unroll
            for (int mt = 0; mt < 2; mt++)
                acc[mt][nt] = __builtin_amdgcn_wmma_f32_16x16x32_f16(
                    false, afr[mt][kc].v, false, bf.v, (short)0, acc[mt][nt], false, false);
        }
    }
    // epilogue: vertical max across tiles (t vs t+4), horizontal via shfl_xor(1),
    // add folded bias, exact GELU, store pooled [b, o, h0, ow]
    int n = lane & 15;
#pragma unroll
    for (int mt = 0; mt < 2; mt++)
#pragma unroll
        for (int t = 0; t < 4; t++)
#pragma unroll
            for (int v = 0; v < 8; v++) {
                float m = fmaxf(acc[mt][t][v], acc[mt][t + 4][v]);
                float other = __shfl_xor(m, 1);
                m = fmaxf(m, other);
                if ((n & 1) == 0) {
                    int o = 32 * wave + 16 * mt + v + 8 * (lane >> 4);
                    float val = m + bp[o];
                    val = 0.5f * val * (1.0f + erff(val * 0.70710678118654752f));
                    int ow = (w0 >> 1) + t * 8 + (n >> 1);
                    out[((size_t)b * NO + o) * (64 * 64) + h0 * 64 + ow] = val;
                }
            }
}

extern "C" void kernel_launch(void* const* d_in, const int* in_sizes, int n_in,
                              void* d_out, int out_size, void* d_ws, size_t ws_size,
                              hipStream_t stream)
{
    const float* x     = (const float*)d_in[0];
    const float* dwh_w = (const float*)d_in[1];
    const float* dwh_b = (const float*)d_in[2];
    const float* dww_w = (const float*)d_in[3];
    const float* dww_b = (const float*)d_in[4];
    const float* wq    = (const float*)d_in[5];
    const float* bq    = (const float*)d_in[6];
    const float* wk    = (const float*)d_in[7];
    const float* bk    = (const float*)d_in[8];
    const float* wv    = (const float*)d_in[9];
    const float* bv    = (const float*)d_in[10];
    const float* wo    = (const float*)d_in[11];
    const float* bo    = (const float*)d_in[12];
    const float* gamma = (const float*)d_in[13];
    const float* beta  = (const float*)d_in[14];
    const float* pw_w  = (const float*)d_in[15];
    const float* pw_b  = (const float*)d_in[16];

    float* ws    = (float*)d_ws;
    float* out   = (float*)d_out;                    // [8,256,64,64]
    float* skip0 = out + (size_t)NB * NO * 64 * 64;  // [8,128,128,128]

    float* dwbuf = ws + OFF_DW;
    float* G     = ws + OFF_G;
    float* bnsum = ws + OFF_BNSUM;
    float* bnsq  = ws + OFF_BNSQ;
    float* sx    = ws + OFF_SX;
    float* Mm    = ws + OFF_M;
    float* cv    = ws + OFF_CV;
    float* Wp    = ws + OFF_WP;
    float* bpv   = ws + OFF_BP;

    k_zero<<<(131328 + 255) / 256, 256, 0, stream>>>(G, 131328); // G + bn sums
    k_axial<<<NB * NC, 256, 0, stream>>>(x, dwh_w, dwh_b, dww_w, dww_b,
                                         dwbuf, bnsum, bnsq, sx);
    k_gram<<<NB * KSPLIT, 256, 0, stream>>>(x, G);
    k_bnfold<<<1, 256, 0, stream>>>(bnsum, bnsq, gamma, beta, pw_w, pw_b, Wp, bpv);
    k_attn<<<NB, 256, 0, stream>>>(G, sx, wq, bq, wk, bk, wv, bv, wo, bo, Mm, cv);
    k_skip0<<<NB * 128, 256, 0, stream>>>(x, Mm, cv, skip0);
    k_pwpool<<<NB * 128, 256, 0, stream>>>(dwbuf, Wp, bpv, out);
}
